// MultiheadAttention_11510512354064
// MI455X (gfx1250) — matrix operand order repaired and checked by reading, not compiled, not run on hardware
//
#include <hip/hip_runtime.h>
#include <hip/hip_bf16.h>

// ---------------------------------------------------------------------------
// MI455X (gfx1250, wave32) multi-head attention:
//   qkv = x @ Wqkv^T + b ; per-head flash attention ; out = vals @ Wo^T + b
// Matrix math via v_wmma_f32_16x16x32_f16; K/V tiles staged with
// global_load_async_to_lds_b128 (ASYNCcnt double buffering).
// ---------------------------------------------------------------------------

typedef __attribute__((ext_vector_type(16))) _Float16 v16h;
typedef __attribute__((ext_vector_type(8)))  float    v8f;

#define BATCH   4
#define SEQ     2048
#define EMBED   1024
#define HEADS   16
#define HDIM    64
#define NQKV    3072           // 3*EMBED
#define ROWS    (BATCH*SEQ)    // 8192

// GEMM LDS tile stride: 36 halves (u32 aligned; 18-dword rows, 18m mod 64
// distinct for m<16 -> conflict-free fragment reads; 72B rows -> 8B aligned)
#define TSTR 36
// Attention LDS tile stride: 72 halves = 144B rows -> 16B aligned for async
// B128 destinations; 36-dword stride: 36n mod 64 is a permutation for n<16.
#define ASTR 72

__device__ inline unsigned int pack2(float a, float b) {
  union { _Float16 h[2]; unsigned int u; } x;
  x.h[0] = (_Float16)a; x.h[1] = (_Float16)b;
  return x.u;
}

// A-matrix fragment (16x32 f16), ISA 7.12.2 layout:
// lane m = lane%16 holds row m; VGPR r pairs at K = (r<4 ? 2r : 2r+8) + 8*hi
__device__ inline v16h load_frag_a(const _Float16* p, int hi) {
  union { v16h h; unsigned int u[8]; } f;
#pragma unroll
  for (int r = 0; r < 8; ++r) {
    int k = ((r < 4) ? (2 * r) : (2 * r + 8)) + 8 * hi;
    f.u[r] = *reinterpret_cast<const unsigned int*>(p + k);
  }
  return f.h;
}

// B-matrix fragment (32x16 f16) read via Bt rows (N x K):
// lane n = lane%16 holds column n; VGPR r pairs at K = (r<4 ? 2r : 2r+8) + 8*hi
__device__ inline v16h load_frag_b(const _Float16* p, int hi) {
  union { v16h h; unsigned int u[8]; } f;
#pragma unroll
  for (int r = 0; r < 8; ++r) {
    int k = ((r < 4) ? (2 * r) : (2 * r + 8)) + 8 * hi;
    f.u[r] = *reinterpret_cast<const unsigned int*>(p + k);
  }
  return f.h;
}

__device__ inline v8f wmma16(v16h a, v16h b, v8f c) {
  return __builtin_amdgcn_wmma_f32_16x16x32_f16(false, a, false, b,
                                                (short)0, c, false, false);
}

// async global -> LDS copy, 16B per active lane; tracked by ASYNCcnt
__device__ inline void async_cp_b128(unsigned lds_off, unsigned long long gaddr) {
  asm volatile("global_load_async_to_lds_b128 %0, %1, off"
               :: "v"(lds_off), "v"(gaddr) : "memory");
}

// cooperative 128x32 tile load, f32 source -> f16 LDS (256 threads)
__device__ inline void load_tile_f32(const float* __restrict__ src, int ld,
                                     _Float16* dst, int tid) {
  int row = tid >> 1;
  int col = (tid & 1) * 16;
  const float4* p = reinterpret_cast<const float4*>(src + (size_t)row * ld + col);
  __builtin_prefetch(reinterpret_cast<const char*>(p) + 128, 0, 0);  // next k-step
  unsigned int* q = reinterpret_cast<unsigned int*>(dst + row * TSTR + col);
#pragma unroll
  for (int i = 0; i < 4; ++i) {
    float4 v = p[i];
    q[2 * i + 0] = pack2(v.x, v.y);
    q[2 * i + 1] = pack2(v.z, v.w);
  }
}

// cooperative 128x32 tile load, f16 source -> f16 LDS (256 threads)
__device__ inline void load_tile_f16(const _Float16* __restrict__ src, int ld,
                                     _Float16* dst, int tid) {
  int row = tid >> 1;
  int col = (tid & 1) * 16;
  const uint4* p = reinterpret_cast<const uint4*>(src + (size_t)row * ld + col);
  __builtin_prefetch(reinterpret_cast<const char*>(p) + 64, 0, 0);   // next k-step
  unsigned int* q = reinterpret_cast<unsigned int*>(dst + row * TSTR + col);
  uint4 v0 = p[0], v1 = p[1];
  q[0] = v0.x; q[1] = v0.y; q[2] = v0.z; q[3] = v0.w;
  q[4] = v1.x; q[5] = v1.y; q[6] = v1.z; q[7] = v1.w;
}

// ---------------------------------------------------------------------------
// Kernel 1: QKV projection.  C(8192x3072) = X(8192x1024) * Wqkv^T + b
// Scatters into Q,K as (b,h,s,d) f16 and V transposed as (b,h,d,s) f16.
// ---------------------------------------------------------------------------
__global__ __launch_bounds__(256) void qkv_gemm_kernel(
    const float* __restrict__ X, const float* __restrict__ W,
    const float* __restrict__ bias, _Float16* __restrict__ Qg,
    _Float16* __restrict__ Kg, _Float16* __restrict__ Vtg) {
  __shared__ _Float16 As[128 * TSTR];
  __shared__ _Float16 Bs[128 * TSTR];

  const int tid  = threadIdx.x;
  const int wave = tid >> 5, lane = tid & 31;
  const int n16  = lane & 15, hi = lane >> 4;
  const int mBase = blockIdx.y * 128, nBase = blockIdx.x * 128;
  const int waveM = (wave >> 1) * 32, waveN = (wave & 1) * 64;

  v8f acc[2][4];
#pragma unroll
  for (int i = 0; i < 2; ++i)
#pragma unroll
    for (int j = 0; j < 4; ++j) acc[i][j] = (v8f){};

  for (int kb = 0; kb < EMBED; kb += 32) {
    __syncthreads();
    load_tile_f32(X + (size_t)mBase * EMBED + kb, EMBED, As, tid);
    load_tile_f32(W + (size_t)nBase * EMBED + kb, EMBED, Bs, tid);
    __syncthreads();
    v16h a[2], b[4];
#pragma unroll
    for (int i = 0; i < 2; ++i)
      a[i] = load_frag_a(&As[(waveM + i * 16 + n16) * TSTR], hi);
#pragma unroll
    for (int j = 0; j < 4; ++j)
      b[j] = load_frag_b(&Bs[(waveN + j * 16 + n16) * TSTR], hi);
#pragma unroll
    for (int i = 0; i < 2; ++i)
#pragma unroll
      for (int j = 0; j < 4; ++j) acc[i][j] = wmma16(a[i], b[j], acc[i][j]);
  }

  // epilogue: bias + scatter into per-head q/k/v (reshape-aware: o -> h=o/192)
#pragma unroll
  for (int i = 0; i < 2; ++i) {
#pragma unroll
    for (int j = 0; j < 4; ++j) {
      const int o  = nBase + waveN + j * 16 + n16;
      const float bv = bias[o];
      const int h = o / 192, c = o % 192;
#pragma unroll
      for (int r = 0; r < 8; ++r) {
        const int gm = mBase + waveM + i * 16 + r + 8 * hi;
        const int bb = gm >> 11, s = gm & (SEQ - 1);
        const int bh = bb * HEADS + h;
        const _Float16 hv = (_Float16)(acc[i][j][r] + bv);
        if (c < 64)
          Qg[((size_t)bh * SEQ + s) * HDIM + c] = hv;
        else if (c < 128)
          Kg[((size_t)bh * SEQ + s) * HDIM + (c - 64)] = hv;
        else
          Vtg[((size_t)bh * HDIM + (c - 128)) * SEQ + s] = hv;
      }
    }
  }
}

// ---------------------------------------------------------------------------
// Kernel 2: flash attention.  One block = 128 queries of one (b,h).
// 8 waves x 16 query rows; K/Vt tiles double-buffered in LDS via
// global_load_async_to_lds_b128; online softmax in f32.
// ---------------------------------------------------------------------------

// stage one 64-key K tile + Vt tile into LDS buffers (4 async issues/wave)
__device__ inline void stage_tiles_async(const _Float16* Kbh,
                                         const _Float16* Vtbh,
                                         _Float16* KsB, _Float16* VtsB,
                                         int kt, int tid) {
  const int row = tid >> 3;        // 0..31
  const int c8  = (tid & 7) * 8;   // halves; 16B chunk per lane
#pragma unroll
  for (int q = 0; q < 2; ++q) {
    const int r = q * 32 + row;
    async_cp_b128(
        (unsigned)(uintptr_t)(KsB + r * ASTR + c8),
        (unsigned long long)(uintptr_t)(Kbh + ((size_t)(kt + r)) * HDIM + c8));
    async_cp_b128(
        (unsigned)(uintptr_t)(VtsB + r * ASTR + c8),
        (unsigned long long)(uintptr_t)(Vtbh + (size_t)r * SEQ + kt + c8));
  }
}

__global__ __launch_bounds__(256) void attn_kernel(
    const _Float16* __restrict__ Qg, const _Float16* __restrict__ Kg,
    const _Float16* __restrict__ Vtg, _Float16* __restrict__ vals) {
  __shared__ _Float16 Ks[2][64 * ASTR];     // 64 keys x 64 d, double buffered
  __shared__ _Float16 Vts[2][64 * ASTR];    // 64 d x 64 keys (V transposed)
  __shared__ _Float16 Ps[8 * 16 * ASTR];    // per-wave P scratch (C->A relayout)

  const int tid  = threadIdx.x;
  const int wave = tid >> 5, lane = tid & 31;
  const int n16  = lane & 15, hi = lane >> 4;
  const int bh   = blockIdx.y;
  const int qRow = blockIdx.x * 128 + wave * 16;

  const _Float16* Kbh  = Kg + (size_t)bh * SEQ * HDIM;
  const _Float16* Vtbh = Vtg + (size_t)bh * HDIM * SEQ;

  // Q A-fragments straight from global (rows contiguous along d)
  const _Float16* qrow = Qg + ((size_t)bh * SEQ + qRow + n16) * HDIM;
  const v16h aq0 = load_frag_a(qrow, hi);
  const v16h aq1 = load_frag_a(qrow + 32, hi);

  v8f acc[4];
  float mrow[8], lrow[8];
#pragma unroll
  for (int j = 0; j < 4; ++j) acc[j] = (v8f){};
#pragma unroll
  for (int r = 0; r < 8; ++r) { mrow[r] = -3.0e38f; lrow[r] = 0.0f; }

  // prologue: stage first tile
  stage_tiles_async(Kbh, Vtbh, Ks[0], Vts[0], 0, tid);
  int buf = 0;

  for (int kt = 0; kt < SEQ; kt += 64) {
    if (kt + 64 < SEQ) {
      // issue next tile into the other buffer, then wait for current tile
      stage_tiles_async(Kbh, Vtbh, Ks[buf ^ 1], Vts[buf ^ 1], kt + 64, tid);
      asm volatile("s_wait_asynccnt 0x4" ::: "memory");  // >4 left = next tile
    } else {
      asm volatile("s_wait_asynccnt 0x0" ::: "memory");
    }
    __syncthreads();  // all waves' current tile visible
    const _Float16* KsB  = Ks[buf];
    const _Float16* VtsB = Vts[buf];

    // S = Q K^T * scale   (4 col-tiles x 2 k-steps)
    v8f sc[4];
#pragma unroll
    for (int j = 0; j < 4; ++j) {
      sc[j] = (v8f){};
      const _Float16* kb = &KsB[(j * 16 + n16) * ASTR];
      sc[j] = wmma16(aq0, load_frag_b(kb, hi), sc[j]);
      sc[j] = wmma16(aq1, load_frag_b(kb + 32, hi), sc[j]);
      sc[j] = sc[j] * 0.125f;  // 1/sqrt(64)
    }

    // online softmax, per output row r (+8*hi); reduce across 16 lanes/half
#pragma unroll
    for (int r = 0; r < 8; ++r) {
      float mx = fmaxf(fmaxf(sc[0][r], sc[1][r]), fmaxf(sc[2][r], sc[3][r]));
      mx = fmaxf(mx, __shfl_xor(mx, 1, 16));
      mx = fmaxf(mx, __shfl_xor(mx, 2, 16));
      mx = fmaxf(mx, __shfl_xor(mx, 4, 16));
      mx = fmaxf(mx, __shfl_xor(mx, 8, 16));
      const float mnew = fmaxf(mrow[r], mx);
      const float corr = __expf(mrow[r] - mnew);
      mrow[r] = mnew;
      lrow[r] *= corr;
#pragma unroll
      for (int j = 0; j < 4; ++j) acc[j][r] *= corr;
      float rs = 0.0f;
#pragma unroll
      for (int j = 0; j < 4; ++j) {
        const float p = __expf(sc[j][r] - mnew);
        sc[j][r] = p;
        rs += p;
      }
      rs += __shfl_xor(rs, 1, 16);
      rs += __shfl_xor(rs, 2, 16);
      rs += __shfl_xor(rs, 4, 16);
      rs += __shfl_xor(rs, 8, 16);
      lrow[r] += rs;
    }

    // relayout P: C-layout regs -> per-wave LDS -> A-layout fragments
    _Float16* pw = Ps + wave * 16 * ASTR;
#pragma unroll
    for (int j = 0; j < 4; ++j)
#pragma unroll
      for (int r = 0; r < 8; ++r)
        pw[(r + 8 * hi) * ASTR + j * 16 + n16] = (_Float16)sc[j][r];
    asm volatile("" ::: "memory");  // keep LDS store->load ordered
    const v16h ap0 = load_frag_a(pw + n16 * ASTR, hi);
    const v16h ap1 = load_frag_a(pw + n16 * ASTR + 32, hi);

    // O += P V   (B-frags from transposed V tile, contiguous rows)
#pragma unroll
    for (int j = 0; j < 4; ++j) {
      const _Float16* vb = &VtsB[(j * 16 + n16) * ASTR];
      acc[j] = wmma16(ap0, load_frag_b(vb, hi), acc[j]);
      acc[j] = wmma16(ap1, load_frag_b(vb + 32, hi), acc[j]);
    }
    __syncthreads();  // everyone done with buf before it is overwritten
    buf ^= 1;
  }

  // normalize and store vals in (b,h,s,d) order == reference's flat reshape
#pragma unroll
  for (int j = 0; j < 4; ++j)
#pragma unroll
    for (int r = 0; r < 8; ++r) {
      const int s = qRow + r + 8 * hi;
      vals[((size_t)bh * SEQ + s) * HDIM + j * 16 + n16] =
          (_Float16)(acc[j][r] * (1.0f / lrow[r]));
    }
}

// ---------------------------------------------------------------------------
// Kernel 3: output projection.  out(8192x1024) = vals * Wo^T + bo   (f32 out)
// ---------------------------------------------------------------------------
__global__ __launch_bounds__(256) void out_gemm_kernel(
    const _Float16* __restrict__ A, const float* __restrict__ W,
    const float* __restrict__ bias, float* __restrict__ out) {
  __shared__ _Float16 As[128 * TSTR];
  __shared__ _Float16 Bs[128 * TSTR];

  const int tid  = threadIdx.x;
  const int wave = tid >> 5, lane = tid & 31;
  const int n16  = lane & 15, hi = lane >> 4;
  const int mBase = blockIdx.y * 128, nBase = blockIdx.x * 128;
  const int waveM = (wave >> 1) * 32, waveN = (wave & 1) * 64;

  v8f acc[2][4];
#pragma unroll
  for (int i = 0; i < 2; ++i)
#pragma unroll
    for (int j = 0; j < 4; ++j) acc[i][j] = (v8f){};

  for (int kb = 0; kb < EMBED; kb += 32) {
    __syncthreads();
    load_tile_f16(A + (size_t)mBase * EMBED + kb, EMBED, As, tid);
    load_tile_f32(W + (size_t)nBase * EMBED + kb, EMBED, Bs, tid);
    __syncthreads();
    v16h a[2], b[4];
#pragma unroll
    for (int i = 0; i < 2; ++i)
      a[i] = load_frag_a(&As[(waveM + i * 16 + n16) * TSTR], hi);
#pragma unroll
    for (int j = 0; j < 4; ++j)
      b[j] = load_frag_b(&Bs[(waveN + j * 16 + n16) * TSTR], hi);
#pragma unroll
    for (int i = 0; i < 2; ++i)
#pragma unroll
      for (int j = 0; j < 4; ++j) acc[i][j] = wmma16(a[i], b[j], acc[i][j]);
  }

#pragma unroll
  for (int i = 0; i < 2; ++i)
#pragma unroll
    for (int j = 0; j < 4; ++j) {
      const int o = nBase + waveN + j * 16 + n16;
      const float bv = bias[o];
#pragma unroll
      for (int r = 0; r < 8; ++r) {
        const int gm = mBase + waveM + i * 16 + r + 8 * hi;
        out[(size_t)gm * EMBED + o] = acc[i][j][r] + bv;
      }
    }
}

// ---------------------------------------------------------------------------
extern "C" void kernel_launch(void* const* d_in, const int* in_sizes, int n_in,
                              void* d_out, int out_size, void* d_ws,
                              size_t ws_size, hipStream_t stream) {
  (void)in_sizes; (void)n_in; (void)out_size; (void)ws_size;
  const float* x     = (const float*)d_in[0];  // (4,2048,1024)
  const float* w_qkv = (const float*)d_in[1];  // (3072,1024)
  const float* b_qkv = (const float*)d_in[2];  // (3072,)
  const float* w_o   = (const float*)d_in[3];  // (1024,1024)
  const float* b_o   = (const float*)d_in[4];  // (1024,)
  float* out = (float*)d_out;                  // (4,2048,1024)

  const size_t nBHSD = (size_t)BATCH * HEADS * SEQ * HDIM;  // 8,388,608
  _Float16* Qg   = (_Float16*)d_ws;            // (b,h,s,d) f16
  _Float16* Kg   = Qg + nBHSD;                 // (b,h,s,d) f16
  _Float16* Vtg  = Kg + nBHSD;                 // (b,h,d,s) f16
  _Float16* vals = Vtg + nBHSD;                // (b,h,s,d) f16  (== (b,s,E) flat)

  qkv_gemm_kernel<<<dim3(NQKV / 128, ROWS / 128), 256, 0, stream>>>(
      x, w_qkv, b_qkv, Qg, Kg, Vtg);
  attn_kernel<<<dim3(SEQ / 128, BATCH * HEADS), 256, 0, stream>>>(
      Qg, Kg, Vtg, vals);
  out_gemm_kernel<<<dim3(EMBED / 128, ROWS / 128), 256, 0, stream>>>(
      vals, w_o, b_o, out);
}
